// CausalSelfAttention_37606733643842
// MI455X (gfx1250) — compile-verified, matrix operand order checked by reading, not run on hardware
//
#include <hip/hip_runtime.h>

typedef unsigned short u16;
typedef __attribute__((ext_vector_type(16))) __bf16 v16bf;
typedef __attribute__((ext_vector_type(8)))  __bf16 v8bf;
typedef __attribute__((ext_vector_type(8)))  float  v8f;
typedef __attribute__((ext_vector_type(4)))  int    v4i;

// ---- problem constants ----
#define PB 2
#define PT 1024
#define PD 2048
#define PN 16
#define PHD 128
#define PK 128
#define PI 4
#define PC 4
#define NEG_BIG (-3.402823466e38f)

// ---- gfx1250 async load-to-LDS (guarded; falls back to reg staging) ----
#if defined(__has_builtin)
# if __has_builtin(__builtin_amdgcn_global_load_async_to_lds_b128)
#  define HAVE_ASYNC_LDS 1
# endif
#endif
#ifndef HAVE_ASYNC_LDS
# define HAVE_ASYNC_LDS 0
#endif
#if HAVE_ASYNC_LDS
typedef __attribute__((address_space(1))) v4i gv4i;   // global int4
typedef __attribute__((address_space(3))) v4i lv4i;   // LDS int4
#endif

__device__ __forceinline__ void wait_async_all() {
#if HAVE_ASYNC_LDS
# if __has_builtin(__builtin_amdgcn_s_wait_asynccnt)
    __builtin_amdgcn_s_wait_asynccnt(0);
# else
    asm volatile("s_wait_asynccnt 0x0" ::: "memory");
# endif
#endif
}

// ---------------- helpers ----------------
__device__ __forceinline__ u16 f2bf(float f) {
    unsigned u = __float_as_uint(f);
    unsigned r = (u + 0x7FFFu + ((u >> 16) & 1u)) >> 16;
    return (u16)r;
}
// load a 16-element bf16 WMMA fragment: two contiguous 16-byte runs at p and p+16 elems
__device__ __forceinline__ v16bf ld_frag(const u16* p) {
    v8bf lo = *(const v8bf*)(p);
    v8bf hi = *(const v8bf*)(p + 16);
    return __builtin_shufflevector(lo, hi, 0,1,2,3,4,5,6,7,8,9,10,11,12,13,14,15);
}
__device__ __forceinline__ float gelu_tanh(float u) {
    return 0.5f * u * (1.f + tanhf(0.7978845608028654f * (u + 0.044715f * u * u * u)));
}

// ---------------- fp32 -> bf16 convert (row major, same layout) ----------------
__global__ void cvt_bf16(const float* __restrict__ src, u16* __restrict__ dst, int n) {
    int i = blockIdx.x * 256 + threadIdx.x;
    if (i < n) dst[i] = f2bf(src[i]);
}

// ---------------- fp32 (R x C) -> bf16 transposed (C x R), tiled through LDS ----------------
__global__ void cvt_bf16_T(const float* __restrict__ src, u16* __restrict__ dst, int R, int Cc) {
    __shared__ float tile[32][33];
    int c0 = blockIdx.x * 32, r0 = blockIdx.y * 32;
    int tx = threadIdx.x, ty = threadIdx.y;            // 32 x 8
    for (int i = ty; i < 32; i += 8)
        tile[i][tx] = src[(size_t)(r0 + i) * Cc + c0 + tx];
    __syncthreads();
    for (int i = ty; i < 32; i += 8)
        dst[(size_t)(c0 + i) * R + r0 + tx] = f2bf(tile[tx][i]);
}

// ---------------- bf16 WMMA GEMM: C[M,N] = A[M,K] @ W[K,N], W given TRANSPOSED (N x K) ----------
// block = 256 threads = 8 waves; wave w: rows m0 = bx*256 + w*32 (2 M-tiles), cols n0 = by*64.
// B tile (64 x 128K) staged in LDS once per block, double-buffered, async if available.
#define BROW 136   // 128 + 8 pad elems (272 B row: kills the 64-bank aliasing of a 256 B stride)
__global__ void gemm_bf16(const u16* __restrict__ A, const u16* __restrict__ WT,
                          float* __restrict__ C, int M, int N, int Kd) {
    __shared__ u16 Bs[2][64][BROW];                    // ~34 KB
    int tid  = threadIdx.x;
    int lane = tid & 31;
    int wave = tid >> 5;
    int m0 = blockIdx.x * 256 + wave * 32;
    int n0 = blockIdx.y * 64;
    int col   = lane & 15;
    int khalf = (lane >> 4) * 8;

    v8f acc[2][4];
    v8f z = {0.f,0.f,0.f,0.f,0.f,0.f,0.f,0.f};
    #pragma unroll
    for (int mt = 0; mt < 2; ++mt)
        #pragma unroll
        for (int nt = 0; nt < 4; ++nt) acc[mt][nt] = z;

    const u16* arow0 = A + (size_t)(m0 + col) * Kd + khalf;
    const u16* arow1 = arow0 + (size_t)16 * Kd;

    // stage one 64x128 B chunk into Bs[buf]: 1024 16-byte segments, 4 per thread
    auto stage = [&](int buf, int k0) {
        #pragma unroll
        for (int ssg = 0; ssg < 4; ++ssg) {
            int seg = tid + ssg * 256;
            int c   = seg >> 4;
            int off = (seg & 15) * 8;
            const u16* g = WT + (size_t)(n0 + c) * Kd + k0 + off;
            u16* l = &Bs[buf][c][off];
#if HAVE_ASYNC_LDS
            __builtin_amdgcn_global_load_async_to_lds_b128((gv4i*)g, (lv4i*)l, 0, 0);
#else
            *(v8bf*)l = *(const v8bf*)g;
#endif
        }
    };

    int NCH = Kd >> 7;                                 // chunks of 128 K
    stage(0, 0);
    for (int ch = 0; ch < NCH; ++ch) {
        wait_async_all();
        __syncthreads();                               // Bs[ch&1] ready; prev buffer fully consumed
        if (ch + 1 < NCH) stage((ch + 1) & 1, (ch + 1) << 7);
        const u16 (*Bb)[BROW] = Bs[ch & 1];
        #pragma unroll
        for (int kk = 0; kk < 128; kk += 32) {
            int k0g = (ch << 7) + kk;
            v16bf a0 = ld_frag(arow0 + k0g);
            v16bf a1 = ld_frag(arow1 + k0g);
            #pragma unroll
            for (int nt = 0; nt < 4; ++nt) {
                v16bf b = ld_frag(&Bb[nt * 16 + col][kk + khalf]);
                acc[0][nt] = __builtin_amdgcn_wmma_f32_16x16x32_bf16(false, a0, false, b, (short)0, acc[0][nt], false, false);
                acc[1][nt] = __builtin_amdgcn_wmma_f32_16x16x32_bf16(false, a1, false, b, (short)0, acc[1][nt], false, false);
            }
        }
    }
    int mhalf = (lane >> 4) * 8;
    #pragma unroll
    for (int mt = 0; mt < 2; ++mt)
        #pragma unroll
        for (int nt = 0; nt < 4; ++nt) {
            int n = n0 + nt * 16 + col;
            if (n >= N) continue;
            #pragma unroll
            for (int r = 0; r < 8; ++r)
                C[(size_t)(m0 + mt * 16 + mhalf + r) * N + n] = acc[mt][nt][r];
        }
}

// ---------------- RoPE (+scale) : (B,T,N,HD) f32 -> (B,N,T,HD) bf16 ----------------
__global__ void rope_kernel(const float* __restrict__ proj, const float* __restrict__ cosb,
                            const float* __restrict__ sinb, u16* __restrict__ out, float scale) {
    int idx = blockIdx.x * 256 + threadIdx.x;          // B*T*N*64 = 2,097,152
    if (idx >= PB * PT * PN * 64) return;
    int h = idx & 63;
    int n = (idx >> 6) & 15;
    int t = (idx >> 10) & 1023;
    int b = idx >> 20;
    const float* u = proj + (((size_t)(b * PT + t) * PN) + n) * PHD;
    float u1 = u[h], u2 = u[h + 64];
    float c = cosb[t * 64 + h], s = sinb[t * 64 + h];
    u16* orow = out + (((size_t)(b * PN + n) * PT) + t) * PHD;
    orow[h]      = f2bf((u1 * c + u2 * s) * scale);
    orow[h + 64] = f2bf((-u1 * s + u2 * c) * scale);
}

// ---------------- V: (B,T,N,HD) f32 -> (B,N,HD,T) bf16 (transposed for PV B-operand) ----------
__global__ void vconv_kernel(const float* __restrict__ vproj, u16* __restrict__ vT) {
    int idx = blockIdx.x * 256 + threadIdx.x;          // t fastest -> coalesced writes
    if (idx >= PB * PN * PHD * PT) return;
    int t = idx & 1023;
    int h = (idx >> 10) & 127;
    int n = (idx >> 17) & 15;
    int b = idx >> 21;
    vT[idx] = f2bf(vproj[(((size_t)(b * PT + t) * PN) + n) * PHD + h]);
}

// ---------------- dynamic weights: gelu(x@dw1)@qkw, rmsnorm, tanh(x@ddw) ----------------
__global__ void dynw_kernel(const float* __restrict__ dwhp, const float* __restrict__ ddp,
                            const float* __restrict__ qkw, float* __restrict__ w1,
                            float* __restrict__ w2, float* __restrict__ ddt) {
    int bt = blockIdx.x;
    int p  = threadIdx.x;
    __shared__ float g[PC * PK];     // 512
    __shared__ float wv[256];
    const float* gp = dwhp + (size_t)bt * (PC * PK);
    for (int i = p; i < PC * PK; i += 256) g[i] = gelu_tanh(gp[i]);
    __syncthreads();

    int c = p >> 6, i_ = (p >> 4) & 3, n = p & 15;
    const float* qp = qkw + c * (PK * PI * PN) + i_ * PN + n;
    float acc = 0.f;
    #pragma unroll 4
    for (int k = 0; k < PK; ++k) acc += g[c * PK + k] * qp[k * (PI * PN)];
    wv[p] = acc;
    __syncthreads();

    if (i_ < 2) {
        int base = p & ~15;
        float s2 = 0.f;
        #pragma unroll
        for (int j = 0; j < 16; ++j) { float t = wv[base + j]; s2 += t * t; }
        float r = rsqrtf(s2 * (1.f / 16.f) + 1e-6f);
        w1[((size_t)bt * PC + c) * 32 + i_ * 16 + n] = acc * r;
    } else {
        w2[((size_t)bt * PC + c) * 32 + (i_ - 2) * 16 + n] = acc;
    }
    if (p < 64) ddt[(size_t)bt * 64 + p] = tanhf(ddp[(size_t)bt * 64 + p]);
}

// ---------------- logits: q@k^T (all heads) + pre cross-head mix + causal mask ----------------
__global__ void logits_kernel(const u16* __restrict__ qb, const u16* __restrict__ kb,
                              const float* __restrict__ sw0, const float* __restrict__ w1,
                              const float* __restrict__ w2, const float* __restrict__ ddt,
                              float* __restrict__ logits) {
    int s0 = blockIdx.x * 16, t0 = blockIdx.y * 16, b = blockIdx.z;
    int tid = threadIdx.x, lane = tid & 31, wave = tid >> 5;

    if (s0 > t0 + 15) {                         // fully-masked tile
        for (int q = tid; q < 16 * 256; q += 512) {
            int m = q >> 8, tt = (q >> 4) & 15, ss = q & 15;
            logits[(((size_t)(b * PN + m) * PT) + t0 + tt) * PT + s0 + ss] = NEG_BIG;
        }
        return;
    }

    __shared__ float Lt[16][16][17];            // [head][t][s] (+pad)
    {
        int col = lane & 15, khalf = (lane >> 4) * 8;
        v8f acc = {0.f,0.f,0.f,0.f,0.f,0.f,0.f,0.f};
        const u16* qrow = qb + ((((size_t)(b * PN + wave) * PT) + t0 + col) * PHD) + khalf;
        const u16* krow = kb + ((((size_t)(b * PN + wave) * PT) + s0 + col) * PHD) + khalf;
        #pragma unroll
        for (int k0 = 0; k0 < PHD; k0 += 32) {
            v16bf a  = ld_frag(qrow + k0);
            v16bf bb = ld_frag(krow + k0);
            acc = __builtin_amdgcn_wmma_f32_16x16x32_bf16(false, a, false, bb, (short)0, acc, false, false);
        }
        int mhalf = (lane >> 4) * 8;
        #pragma unroll
        for (int r = 0; r < 8; ++r) Lt[wave][r + mhalf][col] = acc[r];
    }
    __syncthreads();

    // pre-mix: 2 threads per (t,s) position, 8 output heads each
    int pos = tid >> 1, hh = tid & 1;
    int tt = pos >> 4, ss = pos & 15;
    int t = t0 + tt, s = s0 + ss;
    float Lv[16];
    #pragma unroll
    for (int n = 0; n < 16; ++n) Lv[n] = Lt[n][tt][ss];
    const float* qw1 = w1 + ((size_t)(b * PT + t) * PC + 0) * 32;
    const float* kw1 = w1 + ((size_t)(b * PT + s) * PC + 1) * 32;
    const float* qw2 = w2 + ((size_t)(b * PT + t) * PC + 0) * 32;
    const float* kw2 = w2 + ((size_t)(b * PT + s) * PC + 1) * 32;
    const float* qdd = ddt + (size_t)(b * PT + t) * 64 + 0;
    const float* kdd = ddt + (size_t)(b * PT + s) * 64 + 16;
    float qh0 = 0.f, qh1 = 0.f, kh0 = 0.f, kh1 = 0.f;
    #pragma unroll
    for (int n = 0; n < 16; ++n) {
        qh0 += Lv[n] * qw1[n]; qh1 += Lv[n] * qw1[16 + n];
        kh0 += Lv[n] * kw1[n]; kh1 += Lv[n] * kw1[16 + n];
    }
    for (int m = hh * 8; m < hh * 8 + 8; ++m) {
        float o = Lv[m];
        #pragma unroll
        for (int n = 0; n < 16; ++n) o += Lv[n] * sw0[n * 16 + m];
        o += qh0 * qw2[m] + qh1 * qw2[16 + m] + kh0 * kw2[m] + kh1 * kw2[16 + m];
        o += Lv[m] * (qdd[m] + kdd[m]);
        if (s > t) o = NEG_BIG;
        logits[(((size_t)(b * PN + m) * PT) + t) * PT + s] = o;
    }
}

// ---------------- row softmax in place (row length 1024) ----------------
__global__ void softmax_kernel(float* __restrict__ L) {
    size_t base = (size_t)blockIdx.x * PT;
    int tid = threadIdx.x;
    __shared__ float red[256];
    float v0 = L[base + tid], v1 = L[base + tid + 256], v2 = L[base + tid + 512], v3 = L[base + tid + 768];
    float m = fmaxf(fmaxf(v0, v1), fmaxf(v2, v3));
    red[tid] = m; __syncthreads();
    for (int o = 128; o > 0; o >>= 1) { if (tid < o) red[tid] = fmaxf(red[tid], red[tid + o]); __syncthreads(); }
    m = red[0]; __syncthreads();
    v0 = __expf(v0 - m); v1 = __expf(v1 - m); v2 = __expf(v2 - m); v3 = __expf(v3 - m);
    red[tid] = v0 + v1 + v2 + v3; __syncthreads();
    for (int o = 128; o > 0; o >>= 1) { if (tid < o) red[tid] += red[tid + o]; __syncthreads(); }
    float inv = 1.f / red[0];
    L[base + tid] = v0 * inv; L[base + tid + 256] = v1 * inv;
    L[base + tid + 512] = v2 * inv; L[base + tid + 768] = v3 * inv;
}

// ---------------- post-mix + probs@v : grid (t_tile=64, b=2), block 512 = 16 waves ----------------
// vT layout: (B, N, HD, T) bf16 so B-fragment loads are contiguous along s (the K dim)
__global__ void pv_kernel(const float* __restrict__ probs, const u16* __restrict__ vT,
                          const float* __restrict__ sw1, const float* __restrict__ w1,
                          const float* __restrict__ w2, const float* __restrict__ ddt,
                          u16* __restrict__ ob) {
    int t0 = blockIdx.x * 16, b = blockIdx.y;
    int tid = threadIdx.x, lane = tid & 31, wave = tid >> 5;
    int ntile = blockIdx.x + 1;                 // valid 16-wide s tiles
    int npair = (ntile + 1) >> 1;

    __shared__ float P[16][16][32];             // [head][t][s32]
    __shared__ u16   Am[16][16][40];            // mixed probs bf16, 80 B rows (bank-friendly)

    v8f acc[8];
    v8f z = {0.f,0.f,0.f,0.f,0.f,0.f,0.f,0.f};
    #pragma unroll
    for (int i = 0; i < 8; ++i) acc[i] = z;

    for (int sp = 0; sp < npair; ++sp) {
        int s_base = sp * 32;
        // cooperative load of 16 heads x 16t x 32s probs
        for (int q = 0; q < 16; ++q) {
            int idx = q * 512 + tid;
            int n = idx >> 9, rem = idx & 511, tt = rem >> 5, ss = rem & 31;
            int s = s_base + ss;
            float v = 0.f;
            if (s < ntile * 16)
                v = probs[(((size_t)(b * PN + n) * PT) + t0 + tt) * PT + s];
            P[n][tt][ss] = v;
        }
        __syncthreads();
        // post cross-head mix + mask -> bf16 A fragments
        {
            int tt = tid >> 5, ss = tid & 31;
            int t = t0 + tt, s = s_base + ss;
            float Lv[16];
            #pragma unroll
            for (int n = 0; n < 16; ++n) Lv[n] = P[n][tt][ss];
            const float* qw1 = w1 + ((size_t)(b * PT + t) * PC + 2) * 32;
            const float* qw2 = w2 + ((size_t)(b * PT + t) * PC + 2) * 32;
            const float* kw1 = w1 + ((size_t)(b * PT + s) * PC + 3) * 32;
            const float* kw2 = w2 + ((size_t)(b * PT + s) * PC + 3) * 32;
            const float* qdd = ddt + (size_t)(b * PT + t) * 64 + 32;
            const float* kdd = ddt + (size_t)(b * PT + s) * 64 + 48;
            float qh0 = 0.f, qh1 = 0.f, kh0 = 0.f, kh1 = 0.f;
            #pragma unroll
            for (int n = 0; n < 16; ++n) {
                qh0 += Lv[n] * qw1[n]; qh1 += Lv[n] * qw1[16 + n];
                kh0 += Lv[n] * kw1[n]; kh1 += Lv[n] * kw1[16 + n];
            }
            #pragma unroll
            for (int m = 0; m < 16; ++m) {
                float o = Lv[m];
                #pragma unroll
                for (int n = 0; n < 16; ++n) o += Lv[n] * sw1[n * 16 + m];
                o += qh0 * qw2[m] + qh1 * qw2[16 + m] + kh0 * kw2[m] + kh1 * kw2[16 + m];
                o += Lv[m] * (qdd[m] + kdd[m]);
                if (s > t) o = 0.f;
                Am[m][tt][ss] = f2bf(o);
            }
        }
        __syncthreads();
        // WMMA: o[wave] (16t x 128h) += Am[wave] (16t x 32s) @ v[wave] (32s x 128h)
        {
            int col = lane & 15, khalf = (lane >> 4) * 8;
            v16bf a = ld_frag(&Am[wave][col][khalf]);
            const u16* vbase = vT + (((size_t)(b * PN + wave) * PHD) + col) * PT + s_base + khalf;
            #pragma unroll
            for (int ht = 0; ht < 8; ++ht) {
                v16bf bb = ld_frag(vbase + (size_t)ht * 16 * PT);
                acc[ht] = __builtin_amdgcn_wmma_f32_16x16x32_bf16(false, a, false, bb, (short)0, acc[ht], false, false);
            }
        }
        __syncthreads();
    }
    // write o -> (B,T,N*HD) bf16
    int mhalf = (lane >> 4) * 8, col = lane & 15;
    #pragma unroll
    for (int ht = 0; ht < 8; ++ht)
        #pragma unroll
        for (int r = 0; r < 8; ++r)
            ob[((size_t)(b * PT) + t0 + r + mhalf) * (PN * PHD) + wave * PHD + ht * 16 + col] = f2bf(acc[ht][r]);
}

// ---------------- host launch ----------------
extern "C" void kernel_launch(void* const* d_in, const int* in_sizes, int n_in,
                              void* d_out, int out_size, void* d_ws, size_t ws_size,
                              hipStream_t stream) {
    (void)in_sizes; (void)n_in; (void)out_size; (void)ws_size;
    const float* x    = (const float*)d_in[0];
    const float* wq   = (const float*)d_in[1];
    const float* wk   = (const float*)d_in[2];
    const float* wv   = (const float*)d_in[3];
    const float* wo   = (const float*)d_in[4];
    const float* dw1  = (const float*)d_in[5];
    const float* qkw  = (const float*)d_in[6];
    const float* ddw  = (const float*)d_in[7];
    const float* sw   = (const float*)d_in[8];
    const float* cosb = (const float*)d_in[9];
    const float* sinb = (const float*)d_in[10];
    float* outp = (float*)d_out;

    // workspace carve-up
    char* wsp = (char*)d_ws;
    auto alloc = [&](size_t bytes) -> void* {
        void* p = wsp; wsp += (bytes + 255) & ~(size_t)255; return p;
    };
    const size_t NE_X = (size_t)PB * PT * PD;        // 4,194,304
    u16*   xb    = (u16*)  alloc(NE_X * 2);                       // x, row-major (A operand)
    u16*   wqbT  = (u16*)  alloc((size_t)PD * PN * PHD * 2);      // transposed weights (N x K)
    u16*   wkbT  = (u16*)  alloc((size_t)PD * PN * PHD * 2);
    u16*   wvbT  = (u16*)  alloc((size_t)PD * PN * PHD * 2);
    u16*   wobT  = (u16*)  alloc((size_t)PN * PHD * PD * 2);
    u16*   dw1bT = (u16*)  alloc((size_t)PD * PC * PK * 2);
    u16*   ddwbT = (u16*)  alloc((size_t)PD * PN * PC * 2);
    float* qproj = (float*)alloc(NE_X * 4);
    float* kproj = (float*)alloc(NE_X * 4);
    float* vproj = (float*)alloc(NE_X * 4);
    float* dwhp  = (float*)alloc((size_t)PB * PT * PC * PK * 4);
    float* ddp   = (float*)alloc((size_t)PB * PT * 64 * 4);
    u16*   qb    = (u16*)  alloc(NE_X * 2);
    u16*   kb    = (u16*)  alloc(NE_X * 2);
    u16*   vTb   = (u16*)  alloc(NE_X * 2);
    float* w1    = (float*)alloc((size_t)PB * PT * PC * 32 * 4);
    float* w2    = (float*)alloc((size_t)PB * PT * PC * 32 * 4);
    float* ddt   = (float*)alloc((size_t)PB * PT * 64 * 4);
    float* logit = (float*)alloc((size_t)PB * PN * PT * PT * 4);   // 134 MB, L2-resident
    u16*   ob    = (u16*)  alloc(NE_X * 2);

    auto cdiv = [](int a, int b) { return (a + b - 1) / b; };
    dim3 tb(32, 8);

    // 1) fp32 -> bf16 (x row-major; weights transposed to N x K)
    cvt_bf16<<<cdiv((int)NE_X,256),256,0,stream>>>(x, xb, (int)NE_X);
    cvt_bf16_T<<<dim3((PN*PHD)/32, PD/32), tb, 0, stream>>>(wq,  wqbT,  PD, PN*PHD);
    cvt_bf16_T<<<dim3((PN*PHD)/32, PD/32), tb, 0, stream>>>(wk,  wkbT,  PD, PN*PHD);
    cvt_bf16_T<<<dim3((PN*PHD)/32, PD/32), tb, 0, stream>>>(wv,  wvbT,  PD, PN*PHD);
    cvt_bf16_T<<<dim3(PD/32, (PN*PHD)/32), tb, 0, stream>>>(wo,  wobT,  PN*PHD, PD);
    cvt_bf16_T<<<dim3((PC*PK)/32, PD/32),  tb, 0, stream>>>(dw1, dw1bT, PD, PC*PK);
    cvt_bf16_T<<<dim3((PN*PC)/32, PD/32),  tb, 0, stream>>>(ddw, ddwbT, PD, PN*PC);

    // 2) projections (WMMA GEMMs), grid.x = M/256, grid.y = N/64
    gemm_bf16<<<dim3(8,32),256,0,stream>>>(xb, wqbT, qproj, PB*PT, PN*PHD, PD);
    gemm_bf16<<<dim3(8,32),256,0,stream>>>(xb, wkbT, kproj, PB*PT, PN*PHD, PD);
    gemm_bf16<<<dim3(8,32),256,0,stream>>>(xb, wvbT, vproj, PB*PT, PN*PHD, PD);
    gemm_bf16<<<dim3(8, 8),256,0,stream>>>(xb, dw1bT, dwhp, PB*PT, PC*PK, PD);
    gemm_bf16<<<dim3(8, 1),256,0,stream>>>(xb, ddwbT, ddp,  PB*PT, PN*PC, PD);

    // 3) RoPE + layout
    rope_kernel<<<cdiv(PB*PT*PN*64,256),256,0,stream>>>(qproj, cosb, sinb, qb, 0.08838834764831845f);
    rope_kernel<<<cdiv(PB*PT*PN*64,256),256,0,stream>>>(kproj, cosb, sinb, kb, 1.0f);
    vconv_kernel<<<cdiv((int)NE_X,256),256,0,stream>>>(vproj, vTb);

    // 4) dynamic cross-head weights
    dynw_kernel<<<PB*PT,256,0,stream>>>(dwhp, ddp, qkw, w1, w2, ddt);

    // 5) logits + pre-mix + mask
    logits_kernel<<<dim3(PT/16, PT/16, PB),512,0,stream>>>(qb, kb, sw, w1, w2, ddt, logit);

    // 6) softmax
    softmax_kernel<<<PB*PN*PT,256,0,stream>>>(logit);

    // 7) post-mix + probs@v
    pv_kernel<<<dim3(PT/16, PB),512,0,stream>>>(logit, vTb, sw + PN*PN, w1, w2, ddt, ob);

    // 8) output projection
    gemm_bf16<<<dim3(8,32),256,0,stream>>>(ob, wobT, outp, PB*PT, PD, PD);
}
